// GAT_16638703305092
// MI455X (gfx1250) — compile-verified, matrix operand order checked by reading, not run on hardware
//
#include <hip/hip_runtime.h>
#include <hip/hip_bf16.h>
#include <math.h>

typedef __attribute__((ext_vector_type(16))) __bf16 bf16x16;
typedef __attribute__((ext_vector_type(8)))  float  f32x8;

#define SLOPE 0.2f

// ---------------------------------------------------------------------------
// Dense projection: XW[N,OUTC] = Xin[N,INC] @ W[INC,OUTC].
// One wave computes a 16-row stripe across all column tiles:
//   - A fragments (one per 32-wide K chunk) loaded once, reused for every tile
//   - W staged transposed in LDS as bf16 (stride padded -> conflict-free
//     ds_load_b128 fragment loads), ALL B fragments hoisted before the WMMAs
//   - fully unrolled: KC*CT static v_wmma_f32_16x16x32_bf16 per instantiation
//   - uniform full-tile fast path: no exec divergence around loads/stores
// ---------------------------------------------------------------------------
template<int INC, int OUTC>
__global__ void k_gemm_wmma(const float* __restrict__ Xin,
                            const float* __restrict__ W,
                            float* __restrict__ XW, int N)
{
    constexpr int OUTT = (OUTC + 15) & ~15;   // padded column count
    constexpr int LDST = INC + 8;             // padded bf16 stride per column
    constexpr int KC   = INC / 32;            // K chunks
    constexpr int CT   = OUTT / 16;           // column tiles

    __shared__ __bf16 sWt[OUTT * LDST];       // W transposed: [col][k]
    const int tid = threadIdx.x;
    for (int i = tid; i < OUTT * INC; i += blockDim.x) {
        const int c = i / INC, k = i - c * INC;
        sWt[c * LDST + k] = (c < OUTC) ? (__bf16)W[k * OUTC + c] : (__bf16)0.0f;
    }
    __syncthreads();

    const int wave = tid >> 5, lane = tid & 31;
    const int half = lane >> 4, l16 = lane & 15;   // ISA 16-bit A/B layout
    const int rowBase = (blockIdx.x * (blockDim.x >> 5) + wave) * 16;
    if (rowBase >= N) return;

    // ---- A fragments (per K chunk), reused across all column tiles --------
    bf16x16 a[KC];
    const int  arow = rowBase + l16;
    const bool full = (rowBase + 16 <= N);     // wave-uniform
    if (full || arow < N) {
        const float* pr = Xin + (size_t)arow * INC + half * 8;
#pragma unroll
        for (int kc = 0; kc < KC; ++kc) {
            const float* p = pr + kc * 32;
#pragma unroll
            for (int e = 0; e < 8; ++e) a[kc][e]     = (__bf16)p[e];
#pragma unroll
            for (int e = 0; e < 8; ++e) a[kc][8 + e] = (__bf16)p[16 + e];
        }
    } else {
#pragma unroll
        for (int kc = 0; kc < KC; ++kc)
#pragma unroll
            for (int e = 0; e < 16; ++e) a[kc][e] = (__bf16)0.0f;
    }

    // ---- hoist ALL B fragments: 2*CT*KC ds_load_b128 in flight ------------
    bf16x16 b[CT][KC];
#pragma unroll
    for (int ct = 0; ct < CT; ++ct) {
        const __bf16* colp = sWt + (size_t)(ct * 16 + l16) * LDST + half * 8;
#pragma unroll
        for (int kc = 0; kc < KC; ++kc) {
            const __bf16* q = colp + kc * 32;
#pragma unroll
            for (int e = 0; e < 8; ++e) b[ct][kc][e]     = q[e];
#pragma unroll
            for (int e = 0; e < 8; ++e) b[ct][kc][8 + e] = q[16 + e];
        }
    }

    // ---- back-to-back WMMA chain ------------------------------------------
    f32x8 acc[CT];
#pragma unroll
    for (int ct = 0; ct < CT; ++ct) acc[ct] = (f32x8){};
#pragma unroll
    for (int ct = 0; ct < CT; ++ct)
#pragma unroll
        for (int kc = 0; kc < KC; ++kc)
            acc[ct] = __builtin_amdgcn_wmma_f32_16x16x32_bf16(
                false, a[kc], false, b[ct][kc], (short)0, acc[ct], false, false);

    // ---- store: VGPR v holds row (v + 8*half), column l16 ------------------
    if (full) {                                // uniform branch, no exec churn
#pragma unroll
        for (int ct = 0; ct < CT; ++ct) {
            const int col = ct * 16 + l16;
            if (col < OUTC) {                  // compile-time true for OUTC=64
#pragma unroll
                for (int v = 0; v < 8; ++v)
                    XW[(size_t)(rowBase + v + 8 * half) * OUTC + col] = acc[ct][v];
            }
        }
    } else {
#pragma unroll
        for (int ct = 0; ct < CT; ++ct) {
            const int col = ct * 16 + l16;
#pragma unroll
            for (int v = 0; v < 8; ++v) {
                const int row = rowBase + v + 8 * half;
                if (row < N && col < OUTC)
                    XW[(size_t)row * OUTC + col] = acc[ct][v];
            }
        }
    }
}

// Per-node attention scores: aI[n,h] = sum_c xw[n,h,c]*att[h,c],
//                            aJ[n,h] = sum_c xw[n,h,c]*att[h,C+c]
template<int H, int C>
__global__ void k_scores(const float* __restrict__ XW, const float* __restrict__ att,
                         float* __restrict__ aI, float* __restrict__ aJ, int N)
{
    int idx = blockIdx.x * blockDim.x + threadIdx.x;
    if (idx >= N * H) return;
    int n = idx >> (H == 8 ? 3 : 2), h = idx & (H - 1);
    const float* x  = XW + (size_t)n * H * C + h * C;
    const float* ai = att + (size_t)h * 2 * C;
    float si = 0.f, sj = 0.f;
#pragma unroll
    for (int c = 0; c < C; ++c) { si += x[c] * ai[c]; sj += x[c] * ai[C + c]; }
    aI[idx] = si; aJ[idx] = sj;
}

__global__ void k_init(float* __restrict__ emax, float* __restrict__ den,
                       float* __restrict__ acc, int nh, int nhc)
{
    int i = blockIdx.x * blockDim.x + threadIdx.x;
    if (i < nh) { emax[i] = -INFINITY; den[i] = 0.f; }
    if (i < nhc) acc[i] = 0.f;
}

__device__ __forceinline__ void edge_sd(const long long* __restrict__ ei,
                                        int E, int i, int& s, int& d)
{
    if (i < E) { s = (int)ei[i]; d = (int)ei[(size_t)E + i]; }
    else       { s = i - E; d = i - E; }   // virtual self-loop
}

__device__ __forceinline__ void atomicMaxF(float* addr, float v)
{
    if (v >= 0.f) atomicMax((int*)addr, __float_as_int(v));
    else          atomicMin((unsigned int*)addr, __float_as_uint(v));
}

template<int H>
__device__ __forceinline__ float edge_logit(const float* aI, const float* aJ,
                                            int s, int d, int h)
{
    float v = aI[(size_t)d * H + h] + aJ[(size_t)s * H + h];
    return v > 0.f ? v : SLOPE * v;   // leaky_relu
}

template<int H>
__global__ void k_edge_max(const long long* __restrict__ ei,
                           const float* __restrict__ aI, const float* __restrict__ aJ,
                           float* __restrict__ emax, int E, int EP)
{
    int idx = blockIdx.x * blockDim.x + threadIdx.x;
    if (idx >= EP * H) return;
    int e = idx / H, h = idx & (H - 1);
    int s, d; edge_sd(ei, E, e, s, d);
    atomicMaxF(&emax[(size_t)d * H + h], edge_logit<H>(aI, aJ, s, d, h));
}

template<int H>
__global__ void k_edge_sum(const long long* __restrict__ ei,
                           const float* __restrict__ aI, const float* __restrict__ aJ,
                           const float* __restrict__ emax, float* __restrict__ den,
                           int E, int EP)
{
    int idx = blockIdx.x * blockDim.x + threadIdx.x;
    if (idx >= EP * H) return;
    int e = idx / H, h = idx & (H - 1);
    int s, d; edge_sd(ei, E, e, s, d);
    float v = edge_logit<H>(aI, aJ, s, d, h);
    atomicAdd(&den[(size_t)d * H + h], __expf(v - emax[(size_t)d * H + h]));
}

template<int H, int C>
__global__ void k_edge_scatter(const long long* __restrict__ ei,
                               const float* __restrict__ aI, const float* __restrict__ aJ,
                               const float* __restrict__ emax, const float* __restrict__ den,
                               const float* __restrict__ XW, float* __restrict__ acc,
                               int E, int EP)
{
    int idx = blockIdx.x * blockDim.x + threadIdx.x;
    if (idx >= EP * H) return;
    int e = idx / H, h = idx & (H - 1);
    int s, d; edge_sd(ei, E, e, s, d);
    float v = edge_logit<H>(aI, aJ, s, d, h);
    float alpha = __expf(v - emax[(size_t)d * H + h]) / den[(size_t)d * H + h];
    const float* xs  = XW  + (size_t)s * H * C + h * C;
    float*       out = acc + (size_t)d * H * C + h * C;
#pragma unroll
    for (int c = 0; c < C; ++c)
        atomicAdd(&out[c], xs[c] * alpha);
}

__global__ void k_final_concat(const float* __restrict__ acc, const float* __restrict__ b,
                               float* __restrict__ y, int N, int HC)
{
    int i = blockIdx.x * blockDim.x + threadIdx.x;
    if (i >= N * HC) return;
    float v = acc[i] + b[i & (HC - 1)];
    y[i] = v > 0.f ? v : 0.f;   // relu between layers
}

template<int H>
__global__ void k_final_mean(const float* __restrict__ acc, const float* __restrict__ b,
                             float* __restrict__ y, int N)
{
    int n = blockIdx.x * blockDim.x + threadIdx.x;
    if (n >= N) return;
    float s = 0.f;
#pragma unroll
    for (int h = 0; h < H; ++h) s += acc[(size_t)n * H + h];
    y[n] = s / (float)H + b[0];
}

// ---------------------------------------------------------------------------
extern "C" void kernel_launch(void* const* d_in, const int* in_sizes, int n_in,
                              void* d_out, int out_size, void* d_ws, size_t ws_size,
                              hipStream_t stream)
{
    const float*     X  = (const float*)d_in[0];
    const long long* EI = (const long long*)d_in[1];   // int64 [2, E]
    const int N  = in_sizes[0] / 64;
    const int E  = in_sizes[1] / 2;
    const int EP = E + N;                               // + self-loops

    const float *W1 = (const float*)d_in[2], *att1 = (const float*)d_in[3],
                *b1 = (const float*)d_in[4];
    const float *W2 = (const float*)d_in[5], *att2 = (const float*)d_in[6],
                *b2 = (const float*)d_in[7];
    const float *W3 = (const float*)d_in[8], *att3 = (const float*)d_in[9],
                *b3 = (const float*)d_in[10];

    // workspace carve-out
    char* ws = (char*)d_ws;
    float* XW   = (float*)ws; ws += (size_t)N * 64 * sizeof(float);
    float* ACC  = (float*)ws; ws += (size_t)N * 64 * sizeof(float);
    float* XN   = (float*)ws; ws += (size_t)N * 64 * sizeof(float);
    float* AI   = (float*)ws; ws += (size_t)N * 8  * sizeof(float);
    float* AJ   = (float*)ws; ws += (size_t)N * 8  * sizeof(float);
    float* EMAX = (float*)ws; ws += (size_t)N * 8  * sizeof(float);
    float* DEN  = (float*)ws; ws += (size_t)N * 8  * sizeof(float);

    const int rowTiles = (N + 15) / 16;
    const int gBlocks  = (rowTiles + 7) / 8;            // 8 waves / block (256 thr)
    const int ntBlk8   = (EP * 8 + 255) / 256;
    const int ntBlk4   = (EP * 4 + 255) / 256;

    // ---------------- layer 1: 64 -> (H=8, C=8, concat) --------------------
    k_gemm_wmma<64, 64><<<gBlocks, 256, 0, stream>>>(X, W1, XW, N);
    k_scores<8, 8><<<(N * 8 + 255) / 256, 256, 0, stream>>>(XW, att1, AI, AJ, N);
    k_init<<<(N * 64 + 255) / 256, 256, 0, stream>>>(EMAX, DEN, ACC, N * 8, N * 64);
    k_edge_max<8>    <<<ntBlk8, 256, 0, stream>>>(EI, AI, AJ, EMAX, E, EP);
    k_edge_sum<8>    <<<ntBlk8, 256, 0, stream>>>(EI, AI, AJ, EMAX, DEN, E, EP);
    k_edge_scatter<8, 8><<<ntBlk8, 256, 0, stream>>>(EI, AI, AJ, EMAX, DEN, XW, ACC, E, EP);
    k_final_concat<<<(N * 64 + 255) / 256, 256, 0, stream>>>(ACC, b1, XN, N, 64);

    // ---------------- layer 2: 64 -> (H=8, C=8, concat) --------------------
    k_gemm_wmma<64, 64><<<gBlocks, 256, 0, stream>>>(XN, W2, XW, N);
    k_scores<8, 8><<<(N * 8 + 255) / 256, 256, 0, stream>>>(XW, att2, AI, AJ, N);
    k_init<<<(N * 64 + 255) / 256, 256, 0, stream>>>(EMAX, DEN, ACC, N * 8, N * 64);
    k_edge_max<8>    <<<ntBlk8, 256, 0, stream>>>(EI, AI, AJ, EMAX, E, EP);
    k_edge_sum<8>    <<<ntBlk8, 256, 0, stream>>>(EI, AI, AJ, EMAX, DEN, E, EP);
    k_edge_scatter<8, 8><<<ntBlk8, 256, 0, stream>>>(EI, AI, AJ, EMAX, DEN, XW, ACC, E, EP);
    k_final_concat<<<(N * 64 + 255) / 256, 256, 0, stream>>>(ACC, b2, XN, N, 64);

    // ---------------- layer 3: 64 -> (H=4, C=1, mean) ----------------------
    k_gemm_wmma<64, 4><<<gBlocks, 256, 0, stream>>>(XN, W3, XW, N);
    k_scores<4, 1><<<(N * 4 + 255) / 256, 256, 0, stream>>>(XW, att3, AI, AJ, N);
    k_init<<<(N * 4 + 255) / 256, 256, 0, stream>>>(EMAX, DEN, ACC, N * 4, N * 4);
    k_edge_max<4>    <<<ntBlk4, 256, 0, stream>>>(EI, AI, AJ, EMAX, E, EP);
    k_edge_sum<4>    <<<ntBlk4, 256, 0, stream>>>(EI, AI, AJ, EMAX, DEN, E, EP);
    k_edge_scatter<4, 1><<<ntBlk4, 256, 0, stream>>>(EI, AI, AJ, EMAX, DEN, XW, ACC, E, EP);
    k_final_mean<4><<<(N + 255) / 256, 256, 0, stream>>>(ACC, b3, (float*)d_out, N);
}
